// TDGCN_68822555951719
// MI455X (gfx1250) — matrix operand order repaired and checked by reading, not compiled, not run on hardware
//
#include <hip/hip_runtime.h>
#include <hip/hip_bf16.h>
#include <stdint.h>

#define N_ATOMS 20001
#define KNEI    6
#define FDIM    133
#define CDIM    3
#define HDIM    1100
#define FHDIM   700
#define N_MOLS  800

typedef __bf16 bf16_t;
typedef bf16_t v16bf __attribute__((ext_vector_type(16)));
typedef float  v8f   __attribute__((ext_vector_type(8)));

__device__ __forceinline__ unsigned short f2bf(float f) {
    unsigned u = __builtin_bit_cast(unsigned, f);
    u += 0x7FFFu + ((u >> 16) & 1u);          // round-to-nearest-even
    return (unsigned short)(u >> 16);
}

// ---------------------------------------------------------------- utility
__global__ void k_zero(float* __restrict__ p, long n) {
    long i = (long)blockIdx.x * blockDim.x + threadIdx.x;
    if (i < n) p[i] = 0.0f;
}

// ------------------------------------------------- weight swizzle (f32->bf16)
// Output layout: per (ntG, kt) a 2048-ushort block; inside: tc*512 + lane*16 + elem.
// B-fragment mapping (wave32, 16-bit B 32x16): lane = (n&15) + 16*(kk>>4), elem = kk&15.
__global__ void k_swizzle_W(const float* __restrict__ W, unsigned short* __restrict__ out,
                            int K, int N, int ktiles, long total) {
    long q = (long)blockIdx.x * blockDim.x + threadIdx.x;
    if (q >= total) return;
    int elem = (int)(q & 15);
    int lane = (int)((q >> 4) & 31);
    int tc   = (int)((q >> 9) & 3);
    long rest = q >> 11;                    // = ntG*ktiles + kt
    int kt  = (int)(rest % ktiles);
    int ntG = (int)(rest / ktiles);
    int kk = ((lane >> 4) << 4) + elem;
    int n  = ntG * 64 + tc * 16 + (lane & 15);
    int k  = kt * 32 + kk;
    unsigned short v = 0;
    if (k < K && n < N) v = f2bf(W[(long)k * N + n]);
    out[q] = v;
}

// ---------------------------------------------------------------- WMMA GEMM
// D = [relu]( A(f32,[M,K]) * Bsw(bf16 swizzled,[K,N]) + C0 + bias )
// 256 threads = 8 waves; block tile 128(M) x 64(N); K step 32.
__global__ __launch_bounds__(256) void k_gemm_bf16(
    const float* __restrict__ A, int lda,
    const unsigned short* __restrict__ Bsw,
    const float* __restrict__ C0,
    const float* __restrict__ bias,
    float* __restrict__ outPre,
    float* __restrict__ outRelu,
    int M, int K, int Nout)
{
    __shared__ __align__(32) unsigned short AsF[8 * 32 * 16];  // 8 KB, fragment layout
    __shared__ __align__(32) unsigned short BsF[4 * 32 * 16];  // 4 KB, fragment layout

    const int tid  = threadIdx.x;
    const int wave = tid >> 5;
    const int lane = tid & 31;
    const int l16  = lane & 15;
    const int half = lane >> 4;

    const int rowBase = blockIdx.y * 128;
    const int ntG     = blockIdx.x;
    const int colBase = ntG * 64;
    const int ktiles  = (K + 31) >> 5;

    v8f acc[4] = {};

    // rows >= M only feed output rows that are never stored -> clamp, no predicate
    auto do_tile = [&](int k0, bool ktail) {
        // ---- stage B: 4 KB contiguous copy of pre-swizzled weights ----
        {
            const uint4* srcB = (const uint4*)(Bsw + ((size_t)ntG * ktiles + (k0 >> 5)) * 2048);
            ((uint4*)BsF)[tid] = srcB[tid];
        }
        // ---- stage A tile (128 x 32) into fragment layout, one b128 store/task ----
        // task t: row i = t>>2, K-octet g = t&3.
        // octet g -> lane-half (g>>1), elems (g&1)*8 .. +7  (contiguous!)
        #pragma unroll
        for (int tt = 0; tt < 2; ++tt) {
            int t = tid + tt * 256;
            int i = t >> 2, g = t & 3;
            int grow = rowBase + i;
            const float* arow = A + (long)(grow < M ? grow : M - 1) * lda;
            int kb = k0 + g * 8;
            float f[8];
            #pragma unroll
            for (int l = 0; l < 8; ++l) {
                if (ktail) {
                    int kc = kb + l;
                    float v = arow[kc < K ? kc : K - 1];
                    f[l] = (kc < K) ? v : 0.0f;
                } else {
                    f[l] = arow[kb + l];
                }
            }
            uint4 p;
            p.x = (unsigned)f2bf(f[0]) | ((unsigned)f2bf(f[1]) << 16);
            p.y = (unsigned)f2bf(f[2]) | ((unsigned)f2bf(f[3]) << 16);
            p.z = (unsigned)f2bf(f[4]) | ((unsigned)f2bf(f[5]) << 16);
            p.w = (unsigned)f2bf(f[6]) | ((unsigned)f2bf(f[7]) << 16);
            int w  = i >> 4;
            int ln = (i & 15) + 16 * (g >> 1);
            int eb = (g & 1) * 8;
            *(uint4*)&AsF[(w * 32 + ln) * 16 + eb] = p;
        }
        __syncthreads();

        // ---- fragments: one 32B LDS read each ----
        v16bf af = *(const v16bf*)&AsF[(wave * 32 + lane) * 16];
        #pragma unroll
        for (int tc = 0; tc < 4; ++tc) {
            v16bf bfv = *(const v16bf*)&BsF[(tc * 32 + lane) * 16];
            acc[tc] = __builtin_amdgcn_wmma_f32_16x16x32_bf16(
                false, af, false, bfv, (short)0, acc[tc], false, false);
        }
        __syncthreads();
    };

    const int ktFull = K >> 5;
    for (int kt = 0; kt < ktFull; ++kt) do_tile(kt << 5, false);
    if (K & 31) do_tile(ktFull << 5, true);

    // ---- epilogue ----
    #pragma unroll
    for (int tc = 0; tc < 4; ++tc) {
        int n = colBase + tc * 16 + l16;
        if (n >= Nout) continue;
        float bv = bias ? bias[n] : 0.0f;
        #pragma unroll
        for (int r = 0; r < 8; ++r) {
            int m = rowBase + wave * 16 + r + 8 * half;
            if (m >= M) continue;
            long off = (long)m * Nout + n;
            float val = acc[tc][r] + bv;
            if (C0) val += C0[off];
            if (outPre)  outPre[off]  = val;
            if (outRelu) outRelu[off] = fmaxf(val, 0.0f);
        }
    }
}

// ---------------------------------------------------------------- gathers
__global__ void k_gather_sum(const float* __restrict__ src, const int* __restrict__ a2a,
                             float* __restrict__ out, int rowLen) {
    int n = blockIdx.x;
    int idx[KNEI];
    #pragma unroll
    for (int k = 0; k < KNEI; ++k) idx[k] = a2a[n * KNEI + k];
    for (int j = threadIdx.x; j < rowLen; j += blockDim.x) {
        float acc = 0.0f;
        #pragma unroll
        for (int k = 0; k < KNEI; ++k) acc += src[(long)idx[k] * rowLen + j];
        out[(long)n * rowLen + j] = acc;
    }
}

__global__ void k_gather_sum_norm(const float* __restrict__ src, const int* __restrict__ a2a,
                                  float* __restrict__ out, int rowLen) {
    int n = blockIdx.x;
    int idx[KNEI]; int cnt = 0;
    #pragma unroll
    for (int k = 0; k < KNEI; ++k) { idx[k] = a2a[n * KNEI + k]; cnt += (idx[k] > 0) ? 1 : 0; }
    float inv = 1.0f / (float)(cnt > 0 ? cnt : 1);
    for (int j = threadIdx.x; j < rowLen; j += blockDim.x) {
        float acc = 0.0f;
        #pragma unroll
        for (int k = 0; k < KNEI; ++k) acc += src[(long)idx[k] * rowLen + j];
        out[(long)n * rowLen + j] = acc * inv;
    }
}

// ---------------------------------------------------------------- P = einsum
__global__ void k_build_P(const float* __restrict__ f_atoms, const float* __restrict__ f_bonds,
                          const int* __restrict__ a2a, const int* __restrict__ a2b,
                          float* __restrict__ P) {
    int n = blockIdx.x;
    int idx[KNEI]; float bond[KNEI][CDIM]; float coef[CDIM] = {0.f, 0.f, 0.f};
    #pragma unroll
    for (int k = 0; k < KNEI; ++k) {
        idx[k] = a2a[n * KNEI + k];
        int bi = a2b[n * KNEI + k];
        float msk = (idx[k] > 0) ? 1.0f : 0.0f;
        #pragma unroll
        for (int c = 0; c < CDIM; ++c) {
            bond[k][c] = f_bonds[(long)bi * CDIM + c];
            coef[c] += bond[k][c] * msk;
        }
    }
    for (int f = threadIdx.x; f < FDIM; f += blockDim.x) {
        float fa = f_atoms[(long)n * FDIM + f];
        float nei[KNEI];
        #pragma unroll
        for (int k = 0; k < KNEI; ++k) nei[k] = f_atoms[(long)idx[k] * FDIM + f];
        #pragma unroll
        for (int c = 0; c < CDIM; ++c) {
            float acc = 0.0f;
            #pragma unroll
            for (int k = 0; k < KNEI; ++k) acc += bond[k][c] * nei[k];
            long base = ((long)n * CDIM + c) * (2 * FDIM);
            P[base + f]        = acc;
            P[base + FDIM + f] = coef[c] * fa;
        }
    }
}

// ---------------------------------------------------------------- readout
__global__ void k_segment(const float* __restrict__ os_, const float* __restrict__ ov,
                          const int* __restrict__ mids, float* __restrict__ sums,
                          float* __restrict__ cnt) {
    int n = blockIdx.x + 1;               // skip padding atom 0
    int mid = mids[n];
    for (int j = threadIdx.x; j < FHDIM; j += blockDim.x) {
        atomicAdd(&sums[(long)mid * (2 * FHDIM) + j], os_[(long)n * FHDIM + j]);
        float vv = ov[((long)n * 3 + 0) * FHDIM + j]
                 + ov[((long)n * 3 + 1) * FHDIM + j]
                 + ov[((long)n * 3 + 2) * FHDIM + j];
        atomicAdd(&sums[(long)mid * (2 * FHDIM) + FHDIM + j], vv);
    }
    if (threadIdx.x == 0) atomicAdd(&cnt[mid], 1.0f);
}

__global__ void k_finalize_mol(const float* __restrict__ sums, const float* __restrict__ cnt,
                               float* __restrict__ mol) {
    int i = blockIdx.x * blockDim.x + threadIdx.x;
    if (i < N_MOLS * 2 * FHDIM) {
        int m = i / (2 * FHDIM);
        mol[i] = sums[i] / fmaxf(cnt[m], 1.0f);
    }
}

__global__ void k_final_dot(const float* __restrict__ h, const float* __restrict__ w2,
                            const float* __restrict__ b2, float* __restrict__ out) {
    __shared__ float red[256];
    int m = blockIdx.x;
    float acc = 0.0f;
    for (int j = threadIdx.x; j < FHDIM; j += blockDim.x)
        acc += h[(long)m * FHDIM + j] * w2[j];
    red[threadIdx.x] = acc;
    __syncthreads();
    for (int s = 128; s > 0; s >>= 1) {
        if ((int)threadIdx.x < s) red[threadIdx.x] += red[threadIdx.x + s];
        __syncthreads();
    }
    if (threadIdx.x == 0) out[m] = red[0] + b2[0];
}

// ---------------------------------------------------------------- host
static inline size_t sw_elems(int K, int N) {
    return (size_t)((K + 31) / 32) * (size_t)((N + 63) / 64) * 2048;
}

extern "C" void kernel_launch(void* const* d_in, const int* in_sizes, int n_in,
                              void* d_out, int out_size, void* d_ws, size_t ws_size,
                              hipStream_t stream) {
    (void)in_sizes; (void)n_in; (void)out_size; (void)ws_size;
    const float* f_atoms = (const float*)d_in[0];
    const float* f_bonds = (const float*)d_in[1];
    const float* W_is  = (const float*)d_in[2];
    const float* W_iv  = (const float*)d_in[3];
    const float* W_hs  = (const float*)d_in[4];
    const float* W_hv  = (const float*)d_in[5];
    const float* W_os  = (const float*)d_in[6];
    const float* b_os  = (const float*)d_in[7];
    const float* W_ov  = (const float*)d_in[8];
    const float* b_ov  = (const float*)d_in[9];
    const float* ffn_W1 = (const float*)d_in[10];
    const float* ffn_b1 = (const float*)d_in[11];
    const float* ffn_W2 = (const float*)d_in[12];
    const float* ffn_b2 = (const float*)d_in[13];
    const int* a2a  = (const int*)d_in[14];
    const int* a2b  = (const int*)d_in[15];
    const int* mids = (const int*)d_in[16];

    char* ws = (char*)d_ws;
    size_t off = 0;
    auto alloc = [&](size_t bytes) -> char* {
        char* p = ws + off;
        off = (off + bytes + 255) & ~(size_t)255;
        return p;
    };

    const long NH  = (long)N_ATOMS * HDIM;
    const long NH3 = 3 * NH;
    const long NF  = (long)N_ATOMS * FHDIM;

    float* si  = (float*)alloc(NH  * 4);
    float* s0  = (float*)alloc(NH  * 4);
    float* s1  = (float*)alloc(NH  * 4);
    float* vi  = (float*)alloc(NH3 * 4);
    float* v0  = (float*)alloc(NH3 * 4);
    float* v1  = (float*)alloc(NH3 * 4);
    float* sg  = (float*)alloc(NH  * 4);
    float* vg  = (float*)alloc(NH3 * 4);      // also aliased as P ([3N,266])
    float* tmpOs = (float*)alloc(NF * 4);
    float* osb   = (float*)alloc(NF * 4);
    float* tmpOv = (float*)alloc(3 * NF * 4);
    float* ovb   = (float*)alloc(3 * NF * 4);
    float* sums  = (float*)alloc((size_t)N_MOLS * 2 * FHDIM * 4);
    float* cnt   = (float*)alloc((size_t)N_MOLS * 4);
    float* mol   = (float*)alloc((size_t)N_MOLS * 2 * FHDIM * 4);
    float* hbuf  = (float*)alloc((size_t)N_MOLS * FHDIM * 4);

    unsigned short* bW_is     = (unsigned short*)alloc(sw_elems(FDIM, HDIM) * 2);
    unsigned short* bW_iv     = (unsigned short*)alloc(sw_elems(2 * FDIM, HDIM) * 2);
    unsigned short* bW_hs     = (unsigned short*)alloc(sw_elems(HDIM, HDIM) * 2);
    unsigned short* bW_hv     = (unsigned short*)alloc(sw_elems(HDIM, HDIM) * 2);
    unsigned short* bW_os_top = (unsigned short*)alloc(sw_elems(FDIM, FHDIM) * 2);
    unsigned short* bW_os_bot = (unsigned short*)alloc(sw_elems(HDIM, FHDIM) * 2);
    unsigned short* bW_ov_top = (unsigned short*)alloc(sw_elems(HDIM, FHDIM) * 2);
    unsigned short* bW_ov_bot = (unsigned short*)alloc(sw_elems(HDIM, FHDIM) * 2);
    unsigned short* bW_f1     = (unsigned short*)alloc(sw_elems(2 * FHDIM, FHDIM) * 2);

    auto swz = [&](const float* src, unsigned short* dst, int K, int N) {
        int ktiles = (K + 31) / 32;
        long total = (long)sw_elems(K, N);
        k_swizzle_W<<<dim3((unsigned)((total + 255) / 256)), 256, 0, stream>>>(
            src, dst, K, N, ktiles, total);
    };
    auto gemm = [&](const float* A, int lda, const unsigned short* B,
                    const float* C0, const float* bias,
                    float* outPre, float* outRelu, int M, int K, int Nout) {
        dim3 grid((Nout + 63) / 64, (M + 127) / 128);
        k_gemm_bf16<<<grid, 256, 0, stream>>>(A, lda, B, C0, bias, outPre, outRelu, M, K, Nout);
    };

    // weight conversions to WMMA-fragment-swizzled bf16
    swz(W_is, bW_is, FDIM, HDIM);
    swz(W_iv, bW_iv, 2 * FDIM, HDIM);
    swz(W_hs, bW_hs, HDIM, HDIM);
    swz(W_hv, bW_hv, HDIM, HDIM);
    swz(W_os,                          bW_os_top, FDIM, FHDIM);
    swz(W_os + (size_t)FDIM * FHDIM,   bW_os_bot, HDIM, FHDIM);
    swz(W_ov,                          bW_ov_top, HDIM, FHDIM);
    swz(W_ov + (size_t)HDIM * FHDIM,   bW_ov_bot, HDIM, FHDIM);
    swz(ffn_W1, bW_f1, 2 * FHDIM, FHDIM);

    // stage 1: si = f_atoms @ W_is ; s0 = relu(si)
    gemm(f_atoms, FDIM, bW_is, nullptr, nullptr, si, s0, N_ATOMS, FDIM, HDIM);

    // stage 2: P (neighbor einsum) then vi = P @ W_iv ; v0 = relu(vi)
    float* P = vg;                                  // alias (vg unused yet)
    k_build_P<<<N_ATOMS, 128, 0, stream>>>(f_atoms, f_bonds, a2a, a2b, P);
    gemm(P, 2 * FDIM, bW_iv, nullptr, nullptr, vi, v0, 3 * N_ATOMS, 2 * FDIM, HDIM);

    // message-passing loop
    float* scur = s0; float* soth = s1;
    float* vcur = v0; float* voth = v1;
    for (int it = 0; it < 6; ++it) {
        k_gather_sum<<<N_ATOMS, 256, 0, stream>>>(scur, a2a, sg, HDIM);
        k_gather_sum_norm<<<N_ATOMS, 256, 0, stream>>>(vcur, a2a, vg, 3 * HDIM);
        gemm(sg, HDIM, bW_hs, si, nullptr, nullptr, soth, N_ATOMS, HDIM, HDIM);
        gemm(vg, HDIM, bW_hv, vi, nullptr, nullptr, voth, 3 * N_ATOMS, HDIM, HDIM);
        float* t;
        t = scur; scur = soth; soth = t;
        t = vcur; vcur = voth; voth = t;
    }

    // scalar output head: os = relu([f_atoms | gather(s)] @ W_os + b_os)
    k_gather_sum<<<N_ATOMS, 256, 0, stream>>>(scur, a2a, sg, HDIM);
    gemm(f_atoms, FDIM, bW_os_top, nullptr, b_os, tmpOs, nullptr, N_ATOMS, FDIM, FHDIM);
    gemm(sg, HDIM, bW_os_bot, tmpOs, nullptr, nullptr, osb, N_ATOMS, HDIM, FHDIM);

    // vector output head: ov = relu([vi | gather(v)] @ W_ov + b_ov)
    k_gather_sum<<<N_ATOMS, 256, 0, stream>>>(vcur, a2a, vg, 3 * HDIM);
    gemm(vi, HDIM, bW_ov_top, nullptr, b_ov, tmpOv, nullptr, 3 * N_ATOMS, HDIM, FHDIM);
    gemm(vg, HDIM, bW_ov_bot, tmpOv, nullptr, nullptr, ovb, 3 * N_ATOMS, HDIM, FHDIM);

    // readout: segment mean over real atoms
    long sn = (long)N_MOLS * 2 * FHDIM;
    k_zero<<<dim3((unsigned)((sn + 255) / 256)), 256, 0, stream>>>(sums, sn);
    k_zero<<<dim3((unsigned)((N_MOLS + 255) / 256)), 256, 0, stream>>>(cnt, N_MOLS);
    k_segment<<<N_ATOMS - 1, 256, 0, stream>>>(osb, ovb, mids, sums, cnt);
    k_finalize_mol<<<dim3((unsigned)((sn + 255) / 256)), 256, 0, stream>>>(sums, cnt, mol);

    // FFN
    gemm(mol, 2 * FHDIM, bW_f1, nullptr, ffn_b1, nullptr, hbuf, N_MOLS, 2 * FHDIM, FHDIM);
    k_final_dot<<<N_MOLS, 256, 0, stream>>>(hbuf, ffn_W2, ffn_b2, (float*)d_out);
}
